// TransformerBlock_5368709120668
// MI455X (gfx1250) — matrix operand order repaired and checked by reading, not compiled, not run on hardware
//
#include <hip/hip_runtime.h>
#include <hip/hip_bf16.h>
#include <math.h>

// ---------------- config (matches reference) ----------------
#define CFG_S   2048
#define CFG_H   1024
#define CFG_NH  16
#define CFG_KVH 4
#define CFG_D   64
#define CFG_I   4096
#define CFG_E   8
#define CFG_K   2
#define CFG_W   1024

// ---------------- types ----------------
typedef __bf16 bf16;
typedef __attribute__((ext_vector_type(16))) bf16  v16bf;
typedef __attribute__((ext_vector_type(8)))  float v8f;

// Native f32->bf16 converts (ISel picks packed/scalar v_cvt).
__device__ __forceinline__ bf16 f2bf(float f) { return (bf16)f; }
__device__ __forceinline__ ushort4 f4_to_bf4(float4 v) {
  union { bf16 b[4]; ushort4 u; } o;
  o.b[0] = (bf16)v.x; o.b[1] = (bf16)v.y; o.b[2] = (bf16)v.z; o.b[3] = (bf16)v.w;
  return o.u;
}

// A fragment (16x32 bf16, WMMA A layout): lane L holds row M=L&15,
// K in {half*8..half*8+7} U {16+half*8..16+half*8+7}, half=L>>4.
__device__ __forceinline__ v16bf load_a_frag(const bf16* tile, int stride, int kk) {
  int lane = threadIdx.x & 31;
  const bf16* rp = tile + (size_t)(lane & 15) * stride + kk + ((lane >> 4) << 3);
  union { uint4 u[2]; v16bf v; } u;
  u.u[0] = *(const uint4*)rp;         // K = kk + half*8 .. +7
  u.u[1] = *(const uint4*)(rp + 16);  // K = kk + 16 + half*8 .. +7
  return u.v;
}

// B fragment (32x16 bf16, WMMA B layout) read from an n-major tile
// (tile[n][k], k contiguous): lane L holds col N=n0+(L&15),
// K in {half*8..half*8+7} U {16+half*8..16+half*8+7}, half=L>>4.
__device__ __forceinline__ v16bf load_b_frag(const bf16* tile, int stride, int n0, int kk) {
  int lane = threadIdx.x & 31;
  const bf16* rp = tile + (size_t)(n0 + (lane & 15)) * stride + kk + ((lane >> 4) << 3);
  union { uint4 u[2]; v16bf v; } u;
  u.u[0] = *(const uint4*)rp;
  u.u[1] = *(const uint4*)(rp + 16);
  return u.v;
}

#define WMMA_BF16(A, B, C) \
  __builtin_amdgcn_wmma_f32_16x16x32_bf16(false, (A), false, (B), (short)0, (C), false, false)

// ---------------- LayerNorm ----------------
__global__ void __launch_bounds__(256)
ln_kernel(const float* __restrict__ x, const float* __restrict__ g,
          const float* __restrict__ b, float* __restrict__ y, int Hn) {
  int s = blockIdx.x, tid = threadIdx.x;
  __shared__ float red[256];
  float sum = 0.f, sq = 0.f;
  for (int c = tid; c < Hn; c += 256) {
    float v = x[(size_t)s * Hn + c]; sum += v; sq += v * v;
  }
  red[tid] = sum; __syncthreads();
  for (int o = 128; o > 0; o >>= 1) { if (tid < o) red[tid] += red[tid + o]; __syncthreads(); }
  float mean = red[0] / Hn; __syncthreads();
  red[tid] = sq; __syncthreads();
  for (int o = 128; o > 0; o >>= 1) { if (tid < o) red[tid] += red[tid + o]; __syncthreads(); }
  float var = red[0] / Hn - mean * mean;
  float inv = rsqrtf(var + 1e-5f);
  for (int c = tid; c < Hn; c += 256) {
    float v = x[(size_t)s * Hn + c];
    y[(size_t)s * Hn + c] = (v - mean) * inv * g[c] + b[c];
  }
}

// ---------------- Generic WMMA GEMM (double-buffered LDS) ----------------
// C[M,N] = act(A[M,K] @ B[K,N] + bias) + residual
// Requires: N, M (non-slot path) multiples of 64; Kdim multiple of 64.
// Optional per-expert batching (blockIdx.z) + row gather/scatter via slot lists.
__global__ void __launch_bounds__(128)
gemm_bf16_wmma(const float* __restrict__ A, int lda,
               const float* __restrict__ B, long long bExpStride, int ldb,
               float* __restrict__ C, int ldc,
               int N, int Kdim,
               const float* __restrict__ bias, long long biasExpStride,
               const float* __restrict__ residual, int ldres,
               int act,
               const int* __restrict__ slots, int slotCap,
               const int* __restrict__ counts, int aShift) {
  __shared__ __align__(16) bf16 As[2][64][72];  // padded rows: conflict-free b128 frag loads
  __shared__ __align__(16) bf16 Bt[2][64][72];  // [n][k]
  int e  = blockIdx.z;
  int n0 = blockIdx.x * 64;
  int m0 = blockIdx.y * 64;
  const int* sl = nullptr;
  if (slots) {
    if (m0 >= counts[e]) return;                // uniform early-exit
    sl = slots + (size_t)e * slotCap;
  }
  const float* Bp    = B + (long long)e * bExpStride;
  const float* biasp = bias ? bias + (long long)e * biasExpStride : nullptr;

  int tid = threadIdx.x, wave = tid >> 5, lane = tid & 31;
  int l15 = lane & 15, half = lane >> 4;
  int rbase = tid >> 4;          // staging row base (0..7)
  int c4    = tid & 15;          // staging float4 column (0..15)

  // K-invariant: resolve the 8 A-source rows this thread stages (K-step 64).
  // Padded rows load row 0 unconditionally and are zeroed by a x0 mask.
  const float* aptr[8];
  float amask[8];
  #pragma unroll
  for (int j = 0; j < 8; j++) {
    int srow = m0 + rbase + 8 * j;
    int arow; float msk;
    if (sl) { int pr = sl[srow]; msk = (pr >= 0) ? 1.0f : 0.0f; arow = (pr >= 0) ? (pr >> aShift) : 0; }
    else    { msk = 1.0f; arow = srow; }
    aptr[j]  = A + (size_t)arow * lda;
    amask[j] = msk;
  }

  v8f acc[4];
  #pragma unroll
  for (int t = 0; t < 4; t++) { v8f z = {0,0,0,0,0,0,0,0}; acc[t] = z; }

  // prologue: stage tile k0=0 into buffer 0
  #pragma unroll
  for (int j = 0; j < 8; j++) {
    int r = rbase + 8 * j;
    float4 va = *(const float4*)(aptr[j] + c4 * 4);
    va.x *= amask[j]; va.y *= amask[j]; va.z *= amask[j]; va.w *= amask[j];
    *(ushort4*)&As[0][r][c4 * 4] = f4_to_bf4(va);
    float4 vb = *(const float4*)(Bp + (size_t)r * ldb + n0 + c4 * 4);
    int c = c4 * 4;
    Bt[0][c + 0][r] = f2bf(vb.x);
    Bt[0][c + 1][r] = f2bf(vb.y);
    Bt[0][c + 2][r] = f2bf(vb.z);
    Bt[0][c + 3][r] = f2bf(vb.w);
  }
  __syncthreads();

  for (int k0 = 0; k0 < Kdim; k0 += 64) {
    int cur = (k0 >> 6) & 1, nxt = cur ^ 1;
    bool haveNext = (k0 + 64) < Kdim;

    // issue next tile's global loads first: latency hidden under WMMA compute
    float4 aN[8], bN[8];
    if (haveNext) {
      int kn = k0 + 64;
      if (kn + 64 < Kdim)   // prefetch tile after next (global_prefetch_b8)
        __builtin_prefetch(&Bp[(size_t)(kn + 64 + rbase * 8) * ldb + n0 + c4 * 4], 0, 1);
      #pragma unroll
      for (int j = 0; j < 8; j++) {
        int r = rbase + 8 * j;
        aN[j] = *(const float4*)(aptr[j] + kn + c4 * 4);
        bN[j] = *(const float4*)(Bp + (size_t)(kn + r) * ldb + n0 + c4 * 4);
      }
    }

    // compute on current buffer: all fragment loads up front (disjoint regs),
    // then the 8-WMMA chain with partial dscnt waits
    v16bf a0 = load_a_frag(&As[cur][wave * 16][0], 72, 0);
    v16bf a1 = load_a_frag(&As[cur][wave * 16][0], 72, 32);
    v16bf bf_[8];
    #pragma unroll
    for (int t = 0; t < 4; t++) {
      bf_[2 * t]     = load_b_frag(&Bt[cur][0][0], 72, t * 16, 0);
      bf_[2 * t + 1] = load_b_frag(&Bt[cur][0][0], 72, t * 16, 32);
    }
    #pragma unroll
    for (int t = 0; t < 4; t++) {
      acc[t] = WMMA_BF16(a0, bf_[2 * t],     acc[t]);
      acc[t] = WMMA_BF16(a1, bf_[2 * t + 1], acc[t]);
    }

    // write next tile into the alternate buffer
    if (haveNext) {
      #pragma unroll
      for (int j = 0; j < 8; j++) {
        int r = rbase + 8 * j;
        float4 va = aN[j];
        va.x *= amask[j]; va.y *= amask[j]; va.z *= amask[j]; va.w *= amask[j];
        *(ushort4*)&As[nxt][r][c4 * 4] = f4_to_bf4(va);
        int c = c4 * 4;
        Bt[nxt][c + 0][r] = f2bf(bN[j].x);
        Bt[nxt][c + 1][r] = f2bf(bN[j].y);
        Bt[nxt][c + 2][r] = f2bf(bN[j].z);
        Bt[nxt][c + 3][r] = f2bf(bN[j].w);
      }
    }
    __syncthreads();
  }

  // epilogue: C layout -> lane holds (M = r + 8*half, N = t*16 + l15)
  int crow_[8];
  #pragma unroll
  for (int r = 0; r < 8; r++) {
    int srow = m0 + wave * 16 + r + 8 * half;
    crow_[r] = sl ? sl[srow] : srow;
  }
  #pragma unroll
  for (int t = 0; t < 4; t++) {
    int col = n0 + t * 16 + l15;
    float bv = biasp ? biasp[col] : 0.0f;
    #pragma unroll
    for (int r = 0; r < 8; r++) {
      int crow = crow_[r];
      if (crow >= 0) {
        float v = acc[t][r] + bv;
        if (act == 1) v = 0.5f * v * (1.0f + erff(v * 0.70710678118654752f)); // exact GELU
        if (residual) v += residual[(size_t)crow * ldres + col];
        C[(size_t)crow * ldc + col] = v;
      }
    }
  }
}

// ---------------- Flash attention (sliding window, GQA) ----------------
__global__ void __launch_bounds__(128)
attn_kernel(const float* __restrict__ q, const float* __restrict__ k,
            const float* __restrict__ v, float* __restrict__ o) {
  const int S = CFG_S, H = CFG_H, W = CFG_W, KVS = CFG_KVH * CFG_D;
  int h   = blockIdx.y;
  int q0  = blockIdx.x * 64;
  int kvh = h / (CFG_NH / CFG_KVH);
  __shared__ __align__(16) bf16 Qs[4][16][72];
  __shared__ __align__(16) bf16 Ks[32][72];   // [key][d]
  __shared__ __align__(16) bf16 Vt[64][40];   // [d][key]
  __shared__ __align__(16) bf16 Ps[4][16][40];

  int tid = threadIdx.x, wave = tid >> 5, lane = tid & 31;
  int l15 = lane & 15, half = lane >> 4;
  int rbase = tid >> 4, c4 = tid & 15;

  #pragma unroll
  for (int j = 0; j < 8; j++) {
    int r = rbase + 8 * j;
    float4 vq = *(const float4*)(q + (size_t)(q0 + r) * H + h * 64 + c4 * 4);
    *(ushort4*)&Qs[r >> 4][r & 15][c4 * 4] = f4_to_bf4(vq);
  }
  __syncthreads();
  v16bf aq0 = load_a_frag(&Qs[wave][0][0], 72, 0);
  v16bf aq1 = load_a_frag(&Qs[wave][0][0], 72, 32);

  v8f oacc[4];
  #pragma unroll
  for (int t = 0; t < 4; t++) { v8f z = {0,0,0,0,0,0,0,0}; oacc[t] = z; }
  float mst[8], lst[8];
  #pragma unroll
  for (int r = 0; r < 8; r++) { mst[r] = -INFINITY; lst[r] = 0.0f; }

  int tmax = q0 + 63 + (W - 1);                 // mask: allowed where t <= s + W - 1
  if (tmax > S - 1) tmax = S - 1;
  int nch = tmax / 32 + 1;

  for (int ci = 0; ci < nch; ci++) {
    int t0 = ci * 32;
    #pragma unroll
    for (int j = 0; j < 4; j++) {
      int r = rbase + 8 * j;       // key index within chunk (0..31)
      int t = t0 + r;
      float4 kv = {0.f,0.f,0.f,0.f}, vv = {0.f,0.f,0.f,0.f};
      if (t < S) {
        kv = *(const float4*)(k + (size_t)t * KVS + kvh * 64 + c4 * 4);
        vv = *(const float4*)(v + (size_t)t * KVS + kvh * 64 + c4 * 4);
      }
      *(ushort4*)&Ks[r][c4 * 4] = f4_to_bf4(kv);
      int c = c4 * 4;
      Vt[c + 0][r] = f2bf(vv.x);
      Vt[c + 1][r] = f2bf(vv.y);
      Vt[c + 2][r] = f2bf(vv.z);
      Vt[c + 3][r] = f2bf(vv.w);
    }
    __syncthreads();

    v8f z = {0,0,0,0,0,0,0,0};
    v8f s0 = z, s1 = z;
    s0 = WMMA_BF16(aq0, load_b_frag(&Ks[0][0], 72, 0, 0),   s0);
    s0 = WMMA_BF16(aq1, load_b_frag(&Ks[0][0], 72, 0, 32),  s0);
    s1 = WMMA_BF16(aq0, load_b_frag(&Ks[0][0], 72, 16, 0),  s1);
    s1 = WMMA_BF16(aq1, load_b_frag(&Ks[0][0], 72, 16, 32), s1);

    const float sc2 = 1.0f / (float)CFG_D;      // scale applied twice in reference
    float rmax[8];
    #pragma unroll
    for (int r = 0; r < 8; r++) {
      int srow = q0 + wave * 16 + r + 8 * half;
      int ta = t0 + l15, tb = t0 + 16 + l15;
      float va = s0[r] * sc2, vb = s1[r] * sc2;
      if (ta > srow + W - 1 || ta >= S) va = -INFINITY;
      if (tb > srow + W - 1 || tb >= S) vb = -INFINITY;
      s0[r] = va; s1[r] = vb;
      float m = fmaxf(va, vb);
      m = fmaxf(m, __shfl_xor(m, 1));
      m = fmaxf(m, __shfl_xor(m, 2));
      m = fmaxf(m, __shfl_xor(m, 4));
      m = fmaxf(m, __shfl_xor(m, 8));           // reduce within 16-lane half
      rmax[r] = m;
    }
    #pragma unroll
    for (int r = 0; r < 8; r++) {
      float mn   = fmaxf(mst[r], rmax[r]);
      float corr = __expf(mst[r] - mn);
      float pa = __expf(s0[r] - mn);
      float pb = __expf(s1[r] - mn);
      float ps = pa + pb;
      ps += __shfl_xor(ps, 1);
      ps += __shfl_xor(ps, 2);
      ps += __shfl_xor(ps, 4);
      ps += __shfl_xor(ps, 8);
      lst[r] = lst[r] * corr + ps;
      mst[r] = mn;
      #pragma unroll
      for (int t = 0; t < 4; t++) oacc[t][r] = oacc[t][r] * corr;
      int mrow = r + 8 * half;
      Ps[wave][mrow][l15]      = f2bf(pa);
      Ps[wave][mrow][16 + l15] = f2bf(pb);
    }
    // per-wave LDS round-trip: make P writes visible before A-frag reload
    asm volatile("s_wait_dscnt 0" ::: "memory");
    v16bf pf = load_a_frag(&Ps[wave][0][0], 40, 0);
    #pragma unroll
    for (int t = 0; t < 4; t++)
      oacc[t] = WMMA_BF16(pf, load_b_frag(&Vt[0][0], 40, t * 16, 0), oacc[t]);
    __syncthreads();
  }

  float rinv[8];
  #pragma unroll
  for (int r = 0; r < 8; r++) rinv[r] = 1.0f / lst[r];
  #pragma unroll
  for (int t = 0; t < 4; t++) {
    #pragma unroll
    for (int r = 0; r < 8; r++) {
      int srow = q0 + wave * 16 + r + 8 * half;
      int col  = h * 64 + t * 16 + l15;
      o[(size_t)srow * H + col] = oacc[t][r] * rinv[r];
    }
  }
}

// ---------------- Router: logits, softmax, top-2, renormalize ----------------
__global__ void __launch_bounds__(64)
router_kernel(const float* __restrict__ x2, const float* __restrict__ rw,
              const float* __restrict__ rb, int* __restrict__ tki,
              float* __restrict__ tkp) {
  const int Hn = CFG_H, En = CFG_E;
  int s = blockIdx.x, tid = threadIdx.x;
  float acc[CFG_E];
  #pragma unroll
  for (int e = 0; e < En; e++) acc[e] = 0.0f;
  for (int c = tid; c < Hn; c += 64) {
    float xv = x2[(size_t)s * Hn + c];
    #pragma unroll
    for (int e = 0; e < En; e++) acc[e] += xv * rw[c * En + e];
  }
  __shared__ float red[64][CFG_E];
  #pragma unroll
  for (int e = 0; e < En; e++) red[tid][e] = acc[e];
  __syncthreads();
  for (int o = 32; o > 0; o >>= 1) {
    if (tid < o)
      #pragma unroll
      for (int e = 0; e < En; e++) red[tid][e] += red[tid + o][e];
    __syncthreads();
  }
  if (tid == 0) {
    float lg[CFG_E], mx = -INFINITY, sum = 0.0f;
    #pragma unroll
    for (int e = 0; e < En; e++) { lg[e] = red[0][e] + rb[e]; mx = fmaxf(mx, lg[e]); }
    #pragma unroll
    for (int e = 0; e < En; e++) { lg[e] = __expf(lg[e] - mx); sum += lg[e]; }
    #pragma unroll
    for (int e = 0; e < En; e++) lg[e] /= sum;
    int i0 = 0;
    for (int e = 1; e < En; e++) if (lg[e] > lg[i0]) i0 = e;
    int i1 = (i0 == 0) ? 1 : 0;
    for (int e = 0; e < En; e++) if (e != i0 && lg[e] > lg[i1]) i1 = e;
    float p0 = lg[i0], p1 = lg[i1], ps = p0 + p1;
    tki[2 * s] = i0; tki[2 * s + 1] = i1;
    tkp[2 * s] = p0 / ps; tkp[2 * s + 1] = p1 / ps;
  }
}

// ---------------- MoE routing helpers ----------------
__global__ void moe_init_kernel(int* counts, int* cursors, int* slots, int slotsTotal) {
  int i = blockIdx.x * 256 + threadIdx.x;
  if (i < CFG_E) { counts[i] = 0; cursors[i] = 0; }
  if (i < slotsTotal) slots[i] = -1;
}
__global__ void moe_count_kernel(const int* __restrict__ tki, int* counts, int nPairs) {
  int i = blockIdx.x * 256 + threadIdx.x;
  if (i < nPairs) atomicAdd(&counts[tki[i]], 1);
}
__global__ void moe_scatter_kernel(const int* __restrict__ tki, int* cursors,
                                   int* slots, int nPairs, int cap) {
  int i = blockIdx.x * 256 + threadIdx.x;
  if (i < nPairs) {
    int e = tki[i];
    int pos = atomicAdd(&cursors[e], 1);
    slots[(size_t)e * cap + pos] = i;
  }
}

// ---------------- Combine: out = hidden + p0*y(2s) + p1*y(2s+1) ----------------
__global__ void __launch_bounds__(256)
combine_kernel(const float* __restrict__ hidden, const float* __restrict__ y,
               const float* __restrict__ p, float* __restrict__ out) {
  const int Hn = CFG_H;
  size_t idx = (size_t)blockIdx.x * 256 + threadIdx.x;
  if (idx < (size_t)CFG_S * Hn) {
    int s = (int)(idx / Hn), c = (int)(idx % Hn);
    out[idx] = hidden[idx]
             + p[2 * s]     * y[(size_t)(2 * s)     * Hn + c]
             + p[2 * s + 1] * y[(size_t)(2 * s + 1) * Hn + c];
  }
}

// ---------------- launch ----------------
extern "C" void kernel_launch(void* const* d_in, const int* in_sizes, int n_in,
                              void* d_out, int out_size, void* d_ws, size_t ws_size,
                              hipStream_t stream) {
  (void)in_sizes; (void)n_in; (void)out_size; (void)ws_size;
  const int S = CFG_S, H = CFG_H, I = CFG_I, E = CFG_E, Kk = CFG_K;
  const int KVS = CFG_KVH * CFG_D;

  const float* hs   = (const float*)d_in[0];
  const float* ln1g = (const float*)d_in[1];
  const float* ln1b = (const float*)d_in[2];
  const float* wq   = (const float*)d_in[3];
  const float* wk   = (const float*)d_in[4];
  const float* wv   = (const float*)d_in[5];
  const float* wo   = (const float*)d_in[6];
  const float* ln2g = (const float*)d_in[7];
  const float* ln2b = (const float*)d_in[8];
  const float* rw   = (const float*)d_in[9];
  const float* rb   = (const float*)d_in[10];
  const float* w1   = (const float*)d_in[11];
  const float* b1   = (const float*)d_in[12];
  const float* w2   = (const float*)d_in[13];
  const float* b2   = (const float*)d_in[14];
  float* out = (float*)d_out;

  // workspace carve-up (~131 MB total)
  size_t off = 0;
  char* base = (char*)d_ws;
  auto allocf = [&](size_t n) { float* p = (float*)(base + off); off += n * sizeof(float); return p; };
  float* xln  = allocf((size_t)S * H);
  float* qb   = allocf((size_t)S * H);
  float* kb   = allocf((size_t)S * KVS);
  float* vb   = allocf((size_t)S * KVS);
  float* attn = allocf((size_t)S * H);
  float* hid  = allocf((size_t)S * H);
  float* x2   = allocf((size_t)S * H);
  float* h1   = allocf((size_t)S * Kk * I);
  float* yb   = allocf((size_t)S * Kk * H);
  float* tkp  = allocf((size_t)S * Kk);
  int* tki     = (int*)(base + off); off += (size_t)S * Kk * sizeof(int);
  int* counts  = (int*)(base + off); off += E * sizeof(int);
  int* cursors = (int*)(base + off); off += E * sizeof(int);
  int* slots   = (int*)(base + off); off += (size_t)E * S * Kk * sizeof(int);
  const int slotCap = S * Kk;                 // per-expert capacity (worst case)

  // ---- attention block ----
  ln_kernel<<<S, 256, 0, stream>>>(hs, ln1g, ln1b, xln, H);

  gemm_bf16_wmma<<<dim3(H / 64, S / 64, 1), 128, 0, stream>>>(
      xln, H, wq, 0, H, qb, H, H, H,
      nullptr, 0, nullptr, 0, 0, nullptr, 0, nullptr, 0);
  gemm_bf16_wmma<<<dim3(KVS / 64, S / 64, 1), 128, 0, stream>>>(
      xln, H, wk, 0, KVS, kb, KVS, KVS, H,
      nullptr, 0, nullptr, 0, 0, nullptr, 0, nullptr, 0);
  gemm_bf16_wmma<<<dim3(KVS / 64, S / 64, 1), 128, 0, stream>>>(
      xln, H, wv, 0, KVS, vb, KVS, KVS, H,
      nullptr, 0, nullptr, 0, 0, nullptr, 0, nullptr, 0);

  attn_kernel<<<dim3(S / 64, CFG_NH), 128, 0, stream>>>(qb, kb, vb, attn);

  gemm_bf16_wmma<<<dim3(H / 64, S / 64, 1), 128, 0, stream>>>(
      attn, H, wo, 0, H, hid, H, H, H,
      nullptr, 0, hs, H, 0, nullptr, 0, nullptr, 0);

  // ---- MoE block ----
  ln_kernel<<<S, 256, 0, stream>>>(hid, ln2g, ln2b, x2, H);
  router_kernel<<<S, 64, 0, stream>>>(x2, rw, rb, tki, tkp);

  int slotsTotal = E * slotCap;
  moe_init_kernel<<<(slotsTotal + 255) / 256, 256, 0, stream>>>(counts, cursors, slots, slotsTotal);
  moe_count_kernel<<<(S * Kk + 255) / 256, 256, 0, stream>>>(tki, counts, S * Kk);
  moe_scatter_kernel<<<(S * Kk + 255) / 256, 256, 0, stream>>>(tki, cursors, slots, S * Kk, slotCap);

  // FFN1: h1[pair] = gelu(x2[pair>>1] @ w1[e] + b1[e])
  gemm_bf16_wmma<<<dim3(I / 64, slotCap / 64, E), 128, 0, stream>>>(
      x2, H, w1, (long long)H * I, I, h1, I, I, H,
      b1, (long long)I, nullptr, 0, 1, slots, slotCap, counts, 1);
  // FFN2: y[pair] = h1[pair] @ w2[e] + b2[e]
  gemm_bf16_wmma<<<dim3(H / 64, slotCap / 64, E), 128, 0, stream>>>(
      h1, I, w2, (long long)I * H, H, yb, H, H, I,
      b2, (long long)H, nullptr, 0, 0, slots, slotCap, counts, 0);

  combine_kernel<<<(int)(((size_t)S * H + 255) / 256), 256, 0, stream>>>(hid, yb, tkp, out);
}
